// SwinTransformerBlock_38311108280554
// MI455X (gfx1250) — matrix operand order repaired and checked by reading, compiled, not run on hardware
//
#include <hip/hip_runtime.h>
#include <math.h>

// ---------------------------------------------------------------------------
// Swin-3D block on gfx1250: all GEMMs via v_wmma_f32_16x16x32_f16 (wave32).
// ---------------------------------------------------------------------------

typedef __attribute__((ext_vector_type(16))) _Float16 v16h;
typedef __attribute__((ext_vector_type(8)))  _Float16 v8h;
typedef __attribute__((ext_vector_type(8)))  float    v8f;

#define DIMC   384
#define HEADS  12
#define HD     32          // head dim == one WMMA K step
#define NPAD   112         // 98 tokens padded to 7*16
#define KVPAD  128         // kv dim padded to 4*32 for P@V
#define NWIN   512         // B * 256 windows
#define NTOK   98
#define MROWS  (NWIN * NPAD)       // 57344 window-padded rows
#define TOKENS 50176               // B*D*H*W
#define MLPH   1536

// ---- workspace layout (bytes) ----
static constexpr size_t OFF_WQKV = 0;                         // 384*1152*2
static constexpr size_t OFF_WPROJ = 884736;                   // 384*384*2
static constexpr size_t OFF_WFC1 = 1179648;                   // 384*1536*2
static constexpr size_t OFF_WFC2 = 2359296;                   // 1536*384*2
static constexpr size_t OFF_XW   = 3538944;                   // 57344*384*2  (later reused as attn-out)
static constexpr size_t OFF_Q    = 47579136;                  // 6144*112*32*2 (later reused as LN2 out)
static constexpr size_t OFF_K    = 91619328;                  // 6144*112*32*2
static constexpr size_t OFF_VT   = 135659520;                 // 6144*32*128*2

// ---------------------------------------------------------------------------
// WMMA fragment helpers (contiguous k = 16*(lane/16) + s for A and B, wave32)
// ---------------------------------------------------------------------------
__device__ inline v8f wmma_f16(v16h a, v16h b, v8f c) {
  return __builtin_amdgcn_wmma_f32_16x16x32_f16(false, a, false, b, (short)0, c, false, false);
}

// A (16x32 f16) from row-major [row][k], ld in halves. lane<16: K 0-15; lane>=16: K 16-31
__device__ inline v16h load_a_frag(const _Float16* p, int ld) {
  int lane = threadIdx.x & 31;
  int row  = lane & 15;
  int kb   = (lane >> 4) << 4;
  const _Float16* r = p + (size_t)row * ld + kb;
  v8h lo = *(const v8h*)r;
  v8h hi = *(const v8h*)(r + 8);
  return __builtin_shufflevector(lo, hi, 0,1,2,3,4,5,6,7,8,9,10,11,12,13,14,15);
}

// B (32x16 f16) from [n][k] storage (k contiguous), ld = stride between n rows (halves)
__device__ inline v16h load_b_frag_nk(const _Float16* p, int ld) {
  int lane = threadIdx.x & 31;
  return *(const v16h*)(p + (size_t)(lane & 15) * ld + ((lane >> 4) << 4));
}

// B fragment from pre-packed weight layout: 512 halves per frag, 16 contiguous per lane
__device__ inline v16h load_b_frag_packed(const _Float16* p) {
  return *(const v16h*)(p + ((threadIdx.x & 31) << 4));
}

__device__ inline float wave_sum(float v) {
  #pragma unroll
  for (int o = 16; o > 0; o >>= 1) v += __shfl_xor(v, o, 32);
  return v;
}
__device__ inline float wave_max(float v) {
  #pragma unroll
  for (int o = 16; o > 0; o >>= 1) v = fmaxf(v, __shfl_xor(v, o, 32));
  return v;
}

// ---------------------------------------------------------------------------
// K0: pack fp32 weight (K x N, row-major) into f16 B-fragment layout
// ---------------------------------------------------------------------------
__global__ void pack_b_kernel(const float* __restrict__ w, _Float16* __restrict__ out,
                              int K, int N) {
  int t = blockIdx.x * 256 + threadIdx.x;        // one half-element per thread
  int frag = t >> 9, lane = (t >> 4) & 31, e = t & 15;
  int Ktiles = K >> 5;
  int kt = frag % Ktiles, nt = frag / Ktiles;
  int n = nt * 16 + (lane & 15);
  int k = kt * 32 + ((lane >> 4) << 4) + e;
  out[t] = (_Float16)w[(size_t)k * N + n];
}

__global__ void zero_v8h_kernel(v8h* __restrict__ p) {
  v8h z = {};
  p[(size_t)blockIdx.x * 256 + threadIdx.x] = z;
}

// ---------------------------------------------------------------------------
// K1: LN1 + cyclic shift + window partition -> Xw f16 [NWIN][NPAD][384]
// ---------------------------------------------------------------------------
__global__ void ln1_win_kernel(const float* __restrict__ x, const float* __restrict__ g,
                               const float* __restrict__ bb, _Float16* __restrict__ Xw) {
  int row  = blockIdx.x * 8 + (threadIdx.x >> 5);
  int lane = threadIdx.x & 31;
  int win = row / NPAD, tok = row % NPAD;
  _Float16* dst = Xw + (size_t)row * DIMC;
  if (tok >= NTOK) {                                   // pad rows -> zero
    #pragma unroll
    for (int j = 0; j < 12; j++) dst[lane + 32 * j] = (_Float16)0.0f;
    return;
  }
  int b_ = win >> 8, widx = win & 255;
  int wd = widx >> 6, wh = (widx >> 3) & 7, ww = widx & 7;
  int td = tok / 49, th = (tok / 7) % 7, tw = tok % 7;
  int d = (wd * 2 + td + 1) & 7;                       // roll(-shift)
  int h = (wh * 7 + th + 3) % 56;
  int w = (ww * 7 + tw + 3) % 56;
  const float* src = x + ((((size_t)b_ * 8 + d) * 56 + h) * 56 + w) * DIMC;

  float v[12]; float s = 0.f;
  #pragma unroll
  for (int j = 0; j < 12; j++) { v[j] = src[lane + 32 * j]; s += v[j]; }
  float mu = wave_sum(s) * (1.0f / 384.0f);
  float var = 0.f;
  #pragma unroll
  for (int j = 0; j < 12; j++) { float dd = v[j] - mu; var += dd * dd; }
  var = wave_sum(var) * (1.0f / 384.0f);
  float rstd = rsqrtf(var + 1e-5f);
  #pragma unroll
  for (int j = 0; j < 12; j++) {
    int c = lane + 32 * j;
    dst[c] = (_Float16)((v[j] - mu) * rstd * g[c] + bb[c]);
  }
}

// ---------------------------------------------------------------------------
// K2: QKV GEMM (M=57344,K=384,N=1152) -> Q,K row-major [tok][32], V transposed [32][128]
// ---------------------------------------------------------------------------
__global__ void qkv_gemm_kernel(const _Float16* __restrict__ Xw, const _Float16* __restrict__ Wp,
                                const float* __restrict__ bias, _Float16* __restrict__ Q,
                                _Float16* __restrict__ Kb, _Float16* __restrict__ Vt) {
  int mt = blockIdx.x;
  int wv = threadIdx.x >> 5, lane = threadIdx.x & 31;
  int nt = blockIdx.y * 8 + wv;                        // 0..71
  const _Float16* arow = Xw + (size_t)mt * 16 * DIMC;
  v8f acc = {};
  #pragma unroll
  for (int kt = 0; kt < 12; kt++) {
    v16h a = load_a_frag(arow + kt * 32, DIMC);
    v16h b = load_b_frag_packed(Wp + ((size_t)nt * 12 + kt) * 512);
    acc = wmma_f16(a, b, acc);
  }
  int mloc = (lane >> 4) << 3, nloc = lane & 15;
  int cbase = nt * 16;
  int qkvi = cbase / DIMC;
  int head = (cbase % DIMC) / HD;
  #pragma unroll
  for (int r = 0; r < 8; r++) {
    int grow = mt * 16 + mloc + r;
    int c = cbase + nloc;
    float val = acc[r] + bias[c];
    int win = grow / NPAD, tok = grow % NPAD;
    int dim = c & 31;
    size_t hb = (size_t)(win * HEADS + head);
    if (qkvi == 0)      Q[(hb * NPAD + tok) * HD + dim] = (_Float16)(val * 0.17677669529663687f);
    else if (qkvi == 1) Kb[(hb * NPAD + tok) * HD + dim] = (_Float16)val;
    else                Vt[(hb * HD + dim) * KVPAD + tok] = (_Float16)val;
  }
}

// ---------------------------------------------------------------------------
// K3: attention per (window, head): S=QK^T + bias + mask -> softmax -> P@V^T
// ---------------------------------------------------------------------------
__global__ void attn_kernel(const _Float16* __restrict__ Q, const _Float16* __restrict__ Kb,
                            const _Float16* __restrict__ Vt, const float* __restrict__ btab,
                            const float* __restrict__ mask, _Float16* __restrict__ Out) {
  __shared__ _Float16 Sm[NPAD * KVPAD];                // 28 KB
  int win = blockIdx.x, head = blockIdx.y;
  int wv = threadIdx.x >> 5, lane = threadIdx.x & 31;
  size_t hb = (size_t)(win * HEADS + head);
  const _Float16* q  = Q  + hb * NPAD * HD;
  const _Float16* k  = Kb + hb * NPAD * HD;
  const _Float16* vt = Vt + hb * HD * KVPAD;
  const float* mk = mask + (size_t)(win & 255) * NTOK * NTOK;
  int mloc = (lane >> 4) << 3, nloc = lane & 15;

  // --- stage 1: logits (single WMMA per 16x16 tile: K = head_dim = 32) ---
  for (int t = wv; t < 49; t += 8) {
    int mt = t / 7, nt = t % 7;
    v16h a = load_a_frag(q + (size_t)mt * 16 * HD, HD);
    v16h b = load_b_frag_nk(k + (size_t)nt * 16 * HD, HD);
    v8f acc = {};
    acc = wmma_f16(a, b, acc);
    #pragma unroll
    for (int r = 0; r < 8; r++) {
      int m = mt * 16 + mloc + r;
      int n = nt * 16 + nloc;
      float val;
      if (n >= NTOK) val = -3.0e4f;                    // mask pad kv tokens
      else if (m >= NTOK) val = 0.0f;                  // pad query rows (unused)
      else {
        int td = m / 49, th = (m / 7) % 7, tw = m % 7;
        int nd = n / 49, nh = (n / 7) % 7, nw = n % 7;
        int rel = (td - nd + 1) * 169 + (th - nh + 6) * 13 + (tw - nw + 6);
        val = acc[r] + btab[rel * HEADS + head] + mk[m * NTOK + n];
      }
      Sm[m * KVPAD + n] = (_Float16)val;
    }
  }
  __syncthreads();

  // --- stage 2: softmax over kv dim (wave per row) ---
  for (int rr = 0; rr < 14; rr++) {
    int row = wv * 14 + rr;
    float v[4]; float mx = -3.0e4f;
    #pragma unroll
    for (int j = 0; j < 4; j++) {
      int c = lane + 32 * j;
      v[j] = (c < NPAD) ? (float)Sm[row * KVPAD + c] : -3.0e4f;
      mx = fmaxf(mx, v[j]);
    }
    mx = wave_max(mx);
    float s = 0.f;
    #pragma unroll
    for (int j = 0; j < 4; j++) { v[j] = __expf(v[j] - mx); s += v[j]; }
    float inv = 1.0f / wave_sum(s);
    #pragma unroll
    for (int j = 0; j < 4; j++) Sm[row * KVPAD + lane + 32 * j] = (_Float16)(v[j] * inv);
  }
  __syncthreads();

  // --- stage 3: O = P @ V^T  (A frags from LDS, B frags contiguous from Vt) ---
  for (int t = wv; t < 14; t += 8) {
    int mt = t % 7, ntl = t / 7;
    v8f acc = {};
    #pragma unroll
    for (int kt = 0; kt < 4; kt++) {
      v16h a = load_a_frag(&Sm[mt * 16 * KVPAD + kt * 32], KVPAD);
      v16h b = load_b_frag_nk(vt + (size_t)ntl * 16 * KVPAD + kt * 32, KVPAD);
      acc = wmma_f16(a, b, acc);
    }
    _Float16* op = Out + ((size_t)win * NPAD + mt * 16) * DIMC + head * HD + ntl * 16;
    #pragma unroll
    for (int r = 0; r < 8; r++) op[(size_t)(mloc + r) * DIMC + nloc] = (_Float16)acc[r];
  }
}

// ---------------------------------------------------------------------------
// K4: proj GEMM + window reverse + unshift + shortcut residual -> d_out (f32)
// ---------------------------------------------------------------------------
__global__ void proj_res_kernel(const _Float16* __restrict__ A, const _Float16* __restrict__ Wp,
                                const float* __restrict__ bias, const float* __restrict__ x0,
                                float* __restrict__ Out) {
  int mt = blockIdx.x;
  int wv = threadIdx.x >> 5, lane = threadIdx.x & 31;
  int nt = blockIdx.y * 8 + wv;                        // 0..23
  const _Float16* arow = A + (size_t)mt * 16 * DIMC;
  v8f acc = {};
  #pragma unroll
  for (int kt = 0; kt < 12; kt++) {
    v16h a = load_a_frag(arow + kt * 32, DIMC);
    v16h b = load_b_frag_packed(Wp + ((size_t)nt * 12 + kt) * 512);
    acc = wmma_f16(a, b, acc);
  }
  int mloc = (lane >> 4) << 3, nloc = lane & 15;
  #pragma unroll
  for (int r = 0; r < 8; r++) {
    int grow = mt * 16 + mloc + r;
    int win = grow / NPAD, tok = grow % NPAD;
    if (tok >= NTOK) continue;
    int c = nt * 16 + nloc;
    int b_ = win >> 8, widx = win & 255;
    int wd = widx >> 6, wh = (widx >> 3) & 7, ww = widx & 7;
    int td = tok / 49, th = (tok / 7) % 7, tw = tok % 7;
    int d = (wd * 2 + td + 1) & 7;                     // roll(+shift) of shifted coord
    int h = (wh * 7 + th + 3) % 56;
    int w = (ww * 7 + tw + 3) % 56;
    size_t gi = ((((size_t)b_ * 8 + d) * 56 + h) * 56 + w) * DIMC + c;
    Out[gi] = x0[gi] + acc[r] + bias[c];
  }
}

// ---------------------------------------------------------------------------
// K5: LN2 over d_out -> Hbuf f16 [TOKENS][384]
// ---------------------------------------------------------------------------
__global__ void ln2_kernel(const float* __restrict__ x, const float* __restrict__ g,
                           const float* __restrict__ bb, _Float16* __restrict__ H) {
  int row  = blockIdx.x * 8 + (threadIdx.x >> 5);
  int lane = threadIdx.x & 31;
  const float* src = x + (size_t)row * DIMC;
  _Float16* dst = H + (size_t)row * DIMC;
  float v[12]; float s = 0.f;
  #pragma unroll
  for (int j = 0; j < 12; j++) { v[j] = src[lane + 32 * j]; s += v[j]; }
  float mu = wave_sum(s) * (1.0f / 384.0f);
  float var = 0.f;
  #pragma unroll
  for (int j = 0; j < 12; j++) { float dd = v[j] - mu; var += dd * dd; }
  var = wave_sum(var) * (1.0f / 384.0f);
  float rstd = rsqrtf(var + 1e-5f);
  #pragma unroll
  for (int j = 0; j < 12; j++) {
    int c = lane + 32 * j;
    dst[c] = (_Float16)((v[j] - mu) * rstd * g[c] + bb[c]);
  }
}

// ---------------------------------------------------------------------------
// K6: fused MLP: fc1 + exact GELU into 48KB LDS tile, fc2 accumulated into d_out
// ---------------------------------------------------------------------------
__global__ void mlp_kernel(const _Float16* __restrict__ H, const _Float16* __restrict__ W1,
                           const float* __restrict__ b1, const _Float16* __restrict__ W2,
                           const float* __restrict__ b2, float* __restrict__ Out) {
  __shared__ _Float16 Hh[16 * MLPH];                   // 48 KB hidden tile
  int mt = blockIdx.x;
  int wv = threadIdx.x >> 5, lane = threadIdx.x & 31;
  int mloc = (lane >> 4) << 3, nloc = lane & 15;
  const _Float16* arow = H + (size_t)mt * 16 * DIMC;

  for (int nt = wv; nt < 96; nt += 8) {                // fc1: N=1536
    v8f acc = {};
    #pragma unroll
    for (int kt = 0; kt < 12; kt++) {
      v16h a = load_a_frag(arow + kt * 32, DIMC);
      v16h b = load_b_frag_packed(W1 + ((size_t)nt * 12 + kt) * 512);
      acc = wmma_f16(a, b, acc);
    }
    #pragma unroll
    for (int r = 0; r < 8; r++) {
      int c = nt * 16 + nloc;
      float val = acc[r] + b1[c];
      val = 0.5f * val * (1.0f + erff(val * 0.70710678118654752f));   // exact GELU
      Hh[(mloc + r) * MLPH + c] = (_Float16)val;
    }
  }
  __syncthreads();

  for (int nt = wv; nt < 24; nt += 8) {                // fc2: N=384, K=1536
    v8f acc = {};
    for (int kt = 0; kt < 48; kt++) {
      v16h a = load_a_frag(&Hh[kt * 32], MLPH);        // ds_load path
      v16h b = load_b_frag_packed(W2 + ((size_t)nt * 48 + kt) * 512);
      acc = wmma_f16(a, b, acc);
    }
    #pragma unroll
    for (int r = 0; r < 8; r++) {
      int grow = mt * 16 + mloc + r;
      int c = nt * 16 + nloc;
      Out[(size_t)grow * DIMC + c] += acc[r] + b2[c];
    }
  }
}

// ---------------------------------------------------------------------------
extern "C" void kernel_launch(void* const* d_in, const int* in_sizes, int n_in,
                              void* d_out, int out_size, void* d_ws, size_t ws_size,
                              hipStream_t stream) {
  const float* x     = (const float*)d_in[0];
  const float* mask  = (const float*)d_in[1];
  const float* n1g   = (const float*)d_in[2];
  const float* n1b   = (const float*)d_in[3];
  const float* qkvw  = (const float*)d_in[4];
  const float* qkvb  = (const float*)d_in[5];
  const float* btab  = (const float*)d_in[6];
  const float* projw = (const float*)d_in[7];
  const float* projb = (const float*)d_in[8];
  const float* n2g   = (const float*)d_in[9];
  const float* n2b   = (const float*)d_in[10];
  const float* fc1w  = (const float*)d_in[11];
  const float* fc1b  = (const float*)d_in[12];
  const float* fc2w  = (const float*)d_in[13];
  const float* fc2b  = (const float*)d_in[14];
  float* out = (float*)d_out;
  char* ws = (char*)d_ws;

  _Float16* Wqkv  = (_Float16*)(ws + OFF_WQKV);
  _Float16* Wproj = (_Float16*)(ws + OFF_WPROJ);
  _Float16* Wfc1  = (_Float16*)(ws + OFF_WFC1);
  _Float16* Wfc2  = (_Float16*)(ws + OFF_WFC2);
  _Float16* Xw    = (_Float16*)(ws + OFF_XW);   // reused as attn output after K3
  _Float16* Qb    = (_Float16*)(ws + OFF_Q);    // reused as LN2 output after K3
  _Float16* Kbf   = (_Float16*)(ws + OFF_K);
  _Float16* Vt    = (_Float16*)(ws + OFF_VT);

  pack_b_kernel<<<1728, 256, 0, stream>>>(qkvw,  Wqkv,  384, 1152);
  pack_b_kernel<<<576,  256, 0, stream>>>(projw, Wproj, 384, 384);
  pack_b_kernel<<<2304, 256, 0, stream>>>(fc1w,  Wfc1,  384, 1536);
  pack_b_kernel<<<2304, 256, 0, stream>>>(fc2w,  Wfc2,  1536, 384);
  zero_v8h_kernel<<<12288, 256, 0, stream>>>((v8h*)Vt);        // pad cols of Vt must be 0

  ln1_win_kernel<<<MROWS / 8, 256, 0, stream>>>(x, n1g, n1b, Xw);
  qkv_gemm_kernel<<<dim3(MROWS / 16, 9), 256, 0, stream>>>(Xw, Wqkv, qkvb, Qb, Kbf, Vt);
  attn_kernel<<<dim3(NWIN, HEADS), 256, 0, stream>>>(Qb, Kbf, Vt, btab, mask, Xw);
  proj_res_kernel<<<dim3(MROWS / 16, 3), 256, 0, stream>>>(Xw, Wproj, projb, x, out);
  ln2_kernel<<<TOKENS / 8, 256, 0, stream>>>(out, n2g, n2b, Qb);
  mlp_kernel<<<TOKENS / 16, 256, 0, stream>>>(Qb, Wfc1, fc1b, Wfc2, fc2b, out);
}
